// MHA_26190710571343
// MI455X (gfx1250) — matrix operand order repaired and checked by reading, not compiled, not run on hardware
//
#include <hip/hip_runtime.h>

// ---------------- types ----------------
typedef __attribute__((ext_vector_type(16))) _Float16     v16h;
typedef __attribute__((ext_vector_type(8)))  float        v8f;
typedef __attribute__((ext_vector_type(4)))  unsigned int u32x4;
typedef __attribute__((ext_vector_type(8)))  unsigned int u32x8;
typedef __attribute__((ext_vector_type(4)))  float        f32x4;
typedef __attribute__((ext_vector_type(4)))  _Float16     f16x4;

// ---- CDNA5 async global->LDS copy (probe; fall back to plain b128 copy) ----
#if defined(__has_builtin)
#if __has_builtin(__builtin_amdgcn_global_load_async_to_lds_b128) && \
    __has_builtin(__builtin_amdgcn_s_wait_asynccnt)
#define USE_ASYNC_LDS 1
#endif
#endif
#ifndef USE_ASYNC_LDS
#define USE_ASYNC_LDS 0
#endif

typedef int i4v __attribute__((vector_size(16)));
typedef __attribute__((address_space(1))) i4v* g4p;  // global v4i*
typedef __attribute__((address_space(3))) i4v* l4p;  // LDS v4i*

__device__ __forceinline__ void cp_g2l_16B(_Float16* l, const _Float16* g) {
#if USE_ASYNC_LDS
  __builtin_amdgcn_global_load_async_to_lds_b128((g4p)g, (l4p)l, 0, 0);
#else
  *(u32x4*)l = *(const u32x4*)g;
#endif
}
__device__ __forceinline__ void async_join() {
#if USE_ASYNC_LDS
  __builtin_amdgcn_s_wait_asynccnt(0);
#endif
}

// A-matrix fragment gather (ISA 7.12.2, 16-bit A 16x32):
// lane (m = lane&15, half = lane>>4) reads K bytes [16*half,+16) and [32+16*half,+16)
__device__ __forceinline__ v16h ld_fragA(const _Float16* sm, int row, int pitchBytes,
                                         int kByteOff, int lane) {
  const char* p = (const char*)sm + row * pitchBytes + kByteOff + ((lane >> 4) << 4);
  u32x4 a = *(const u32x4*)(p);
  u32x4 b = *(const u32x4*)(p + 32);
  u32x8 c = {a[0], a[1], a[2], a[3], b[0], b[1], b[2], b[3]};
  return __builtin_bit_cast(v16h, c);
}
// B-matrix fragment gather (lane = N column): bytes [16*half,+16) and [32+16*half,+16) of transposed-B row
__device__ __forceinline__ v16h ld_fragB(const _Float16* sm, int row, int pitchBytes,
                                         int kByteOff, int lane) {
  const char* p = (const char*)sm + row * pitchBytes + kByteOff + ((lane >> 4) << 4);
  u32x4 a = *(const u32x4*)(p);
  u32x4 b = *(const u32x4*)(p + 32);
  u32x8 c = {a[0], a[1], a[2], a[3], b[0], b[1], b[2], b[3]};
  return __builtin_bit_cast(v16h, c);
}

// ---------------- pre-passes ----------------
__global__ __launch_bounds__(256) void cvt_f32_f16(const float* __restrict__ s,
                                                   _Float16* __restrict__ d, int n4) {
  int i = blockIdx.x * 256 + threadIdx.x;
  if (i < n4) {
    f32x4 v = *(const f32x4*)(s + (long long)i * 4);
    f16x4 h = {(_Float16)v[0], (_Float16)v[1], (_Float16)v[2], (_Float16)v[3]};
    *(f16x4*)(d + (long long)i * 4) = h;
  }
}

// dst[c][r] = (f16) src[r][c];  src: [R][Cdim] fp32, dst: [Cdim][R] f16
__global__ __launch_bounds__(256) void transpose_cvt(
    const float* __restrict__ src, _Float16* __restrict__ dst,
    int R, int Cdim, long long sSrc, long long sDst)
{
  __shared__ float tile[32][33];
  const float* s = src + (long long)blockIdx.z * sSrc;
  _Float16*   d = dst + (long long)blockIdx.z * sDst;
  const int rt = blockIdx.x * 32, ct = blockIdx.y * 32;
  const int t = threadIdx.x;
  {
    int r = t >> 3, g = t & 7;
    f32x4 v = *(const f32x4*)(s + (long long)(rt + r) * Cdim + ct + g * 4);
    tile[r][g * 4 + 0] = v[0]; tile[r][g * 4 + 1] = v[1];
    tile[r][g * 4 + 2] = v[2]; tile[r][g * 4 + 3] = v[3];
  }
  __syncthreads();
  {
    int c = t >> 3, rg = t & 7;
    f16x4 h = {(_Float16)tile[rg * 4 + 0][c], (_Float16)tile[rg * 4 + 1][c],
               (_Float16)tile[rg * 4 + 2][c], (_Float16)tile[rg * 4 + 3][c]};
    *(f16x4*)(d + (long long)(ct + c) * R + rt + rg * 4) = h;
  }
}

// ---------------- f16-WMMA GEMM, double-buffered LDS ----------------
// C = A[M,K] @ B[K,N] (+bias). BT is pre-transposed f16 [N][K].
// A16: A is f16 [M][K]; else fp32 (converted while staging). OUT16: C stored f16.
// Tile 64(M) x 256(N) x 32(K); 256 threads = 8 waves as 2(M) x 4(N).
template <bool A16, bool OUT16>
__global__ __launch_bounds__(256) void wmma_gemm(
    const void* __restrict__ Ap, const _Float16* __restrict__ BT,
    const float* __restrict__ bias, void* __restrict__ Cp,
    int Kdim, int N,
    long long strideA, long long strideB, long long strideC)
{
  __shared__ __align__(16) _Float16 smA[2][64 * 40];   // pitch 40 halfs (80 B)
  __shared__ __align__(16) _Float16 smB[2][256 * 40];  // [n][k] transposed

  const _Float16* Bb = BT + (long long)blockIdx.z * strideB;
  const int t = threadIdx.x;
  const int lane = t & 31, wave = t >> 5;
  const int wm = wave >> 2, wn = wave & 3;
  const long long m0 = (long long)blockIdx.y * 64;
  const int n0 = blockIdx.x * 256;

  const v8f vzero = {0.f, 0.f, 0.f, 0.f, 0.f, 0.f, 0.f, 0.f};
  v8f acc[2][4];
  for (int i = 0; i < 2; ++i)
    for (int j = 0; j < 4; ++j) acc[i][j] = vzero;

  const int kTiles = Kdim >> 5;
  const int gC = t & 3, rC = t >> 2;  // 16B-chunk staging coords

  auto stageB = [&](int kt, int buf) {
    for (int p = 0; p < 4; ++p) {
      int row = rC + p * 64;
      cp_g2l_16B(&smB[buf][row * 40 + gC * 8],
                 Bb + (long long)(n0 + row) * Kdim + kt * 32 + gC * 8);
    }
  };
  auto stageA = [&](int kt, int buf) {
    if (A16) {
      const _Float16* Af = (const _Float16*)Ap + (long long)blockIdx.z * strideA;
      cp_g2l_16B(&smA[buf][rC * 40 + gC * 8],
                 Af + (m0 + rC) * (long long)Kdim + kt * 32 + gC * 8);
    } else {
      const float* Af = (const float*)Ap + (long long)blockIdx.z * strideA;
      int kq = t & 7, mA = t >> 3;
      for (int p = 0; p < 2; ++p) {
        int m = mA + p * 32;
        f32x4 v = *(const f32x4*)(Af + (m0 + m) * (long long)Kdim + kt * 32 + kq * 4);
        f16x4 h = {(_Float16)v[0], (_Float16)v[1], (_Float16)v[2], (_Float16)v[3]};
        *(f16x4*)(&smA[buf][m * 40 + kq * 4]) = h;
      }
    }
  };

  stageA(0, 0);
  stageB(0, 0);
  for (int kt = 0; kt < kTiles; ++kt) {
    const int buf = kt & 1;
    async_join();        // current tile's async copies complete
    __syncthreads();     // ...and visible; everyone done reading other buffer
    if (kt + 1 < kTiles) {  // prefetch next tile, overlapped with WMMA below
      stageA(kt + 1, buf ^ 1);
      stageB(kt + 1, buf ^ 1);
    }
    v16h af[2], bfv[4];
    for (int i = 0; i < 2; ++i)
      af[i] = ld_fragA(smA[buf], wm * 32 + i * 16 + (lane & 15), 80, 0, lane);
    for (int j = 0; j < 4; ++j)
      bfv[j] = ld_fragB(smB[buf], wn * 64 + j * 16 + (lane & 15), 80, 0, lane);
    for (int i = 0; i < 2; ++i)
      for (int j = 0; j < 4; ++j)
        acc[i][j] = __builtin_amdgcn_wmma_f32_16x16x32_f16(
            false, af[i], false, bfv[j], (short)0, acc[i][j], false, false);
  }

  // epilogue: C/D layout -> lane: m = r + 8*(lane>=16), n = lane&15
  const int half = lane >> 4, c16 = lane & 15;
  for (int i = 0; i < 2; ++i)
    for (int j = 0; j < 4; ++j) {
      int ncol = n0 + wn * 64 + j * 16 + c16;
      float bvv = bias ? bias[ncol] : 0.f;
      for (int r = 0; r < 8; ++r) {
        long long row = m0 + wm * 32 + i * 16 + r + 8 * half;
        float val = acc[i][j][r] + bvv;
        if (OUT16)
          ((_Float16*)Cp + (long long)blockIdx.z * strideC)[row * (long long)N + ncol] =
              (_Float16)val;
        else
          ((float*)Cp + (long long)blockIdx.z * strideC)[row * (long long)N + ncol] = val;
      }
    }
}

// ---------------- chunk stats (rf_k_bar, beta_lm) ----------------
__device__ __forceinline__ float block_reduce_sum64(float x, int d, float* sR) {
  __syncthreads();
  sR[d] = x;
  __syncthreads();
  for (int s = 32; s > 0; s >>= 1) {
    if (d < s) sR[d] += sR[d + s];
    __syncthreads();
  }
  return sR[0];
}

__device__ __forceinline__ float dense_ln64(float x, const float* __restrict__ W,
    const float* __restrict__ bb, const float* __restrict__ g, const float* __restrict__ be,
    int d, float* sX, float* sR) {
  __syncthreads();
  sX[d] = x;
  __syncthreads();
  float y = bb[d];
  for (int e = 0; e < 64; ++e) y += sX[e] * W[e * 64 + d];
  float mean = block_reduce_sum64(y, d, sR) * (1.f / 64.f);
  float tt = y - mean;
  float var = block_reduce_sum64(tt * tt, d, sR) * (1.f / 64.f);
  return tt * rsqrtf(var + 1e-6f) * g[d] + be[d];
}

// one block (64 threads) per (b,h,c); thread owns one d
__global__ __launch_bounds__(64) void chunk_stats(
    const _Float16* __restrict__ q, const float* __restrict__ kp, const float* __restrict__ vp,
    const float* __restrict__ mqW, const float* __restrict__ mqb,
    const float* __restrict__ mqg, const float* __restrict__ mqbe,
    const float* __restrict__ mkW, const float* __restrict__ mkb,
    const float* __restrict__ mkg, const float* __restrict__ mkbe,
    float* __restrict__ kbar, float* __restrict__ beta)
{
  __shared__ float sX[64], sR[64];
  const int bc = blockIdx.x;
  const int c = bc & 31, h = (bc >> 5) % 12, b = bc / 384;
  const int d = threadIdx.x;

  const _Float16* qrow = q + ((long long)b * 1024 + c * 32) * 768 + h * 64 + d;
  float qm = 0.f;
  for (int i = 0; i < 32; ++i) qm += (float)qrow[i * 768];
  qm *= (1.f / 32.f);
  const float* krow = kp + ((long long)b * 64 + c * 2) * 768 + h * 64 + d;
  float k0 = krow[0], k1 = krow[768];
  float km = 0.5f * (k0 + k1);

  float qb_ = dense_ln64(qm, mqW, mqb, mqg, mqbe, d, sX, sR);
  float kb_ = dense_ln64(km, mkW, mkb, mkg, mkbe, d, sX, sR);
  float mu = 0.5f * (qb_ + kb_);

  const long long o = ((long long)(b * 12 + h) * 32 + c) * 64 + d;
  kbar[o] = kb_;

  float dot0 = block_reduce_sum64(mu * k0, d, sR);
  float ss0  = block_reduce_sum64(k0 * k0, d, sR);
  float dot1 = block_reduce_sum64(mu * k1, d, sR);
  float ss1  = block_reduce_sum64(k1 * k1, d, sR);
  const float dn = 0.125f;  // DH^-0.5
  float lp0 = dn * dot0 - 0.5f * dn * ss0;
  float lp1 = dn * dot1 - 0.5f * dn * ss1;
  float mx = fmaxf(lp0, lp1);
  float e0 = __expf(lp0 - mx), e1 = __expf(lp1 - mx);
  float inv = 1.f / (e0 + e1);
  const float* vrow = vp + ((long long)b * 64 + c * 2) * 768 + h * 64 + d;
  beta[o] = e0 * inv * vrow[0] + e1 * inv * vrow[768];
}

// ---------------- windowed attention (in-place on f16 q buffer) ----------------
__global__ __launch_bounds__(128) void window_attn(
    const float* __restrict__ kp, const float* __restrict__ vp,
    const float* __restrict__ kbar, const float* __restrict__ beta,
    _Float16* __restrict__ qio)
{
  __shared__ __align__(16) _Float16 sQ[32 * 72];   // [i][d], pitch 72 halfs (144 B)
  __shared__ __align__(16) _Float16 sKB[32 * 72];  // [c][d] == B^T for Q@Kbar^T
  __shared__ __align__(16) _Float16 sRA[32 * 40];  // ra probs [i][c]
  __shared__ __align__(16) _Float16 sBT[64 * 40];  // beta^T [d][c]
  __shared__ float sKl[2 * 64], sVl[2 * 64];
  __shared__ float sS[32 * 36];                    // cols 0..1 local, 2..33 rfa

  const int bc = blockIdx.x;
  const int w = bc & 31, h = (bc >> 5) % 12, b = bc / 384;
  const int t = threadIdx.x, lane = t & 31, wv = t >> 5;
  const long long qrow0 = (long long)b * 1024 + w * 32;

  for (int p = 0; p < 2; ++p) {           // copy f16 Q tile, 16B chunks
    int cidx = p * 128 + t;
    int i = cidx >> 3, g = cidx & 7;
    *(u32x4*)&sQ[i * 72 + g * 8] =
        *(const u32x4*)(qio + (qrow0 + i) * 768 + h * 64 + g * 8);
  }
  const float* kbr = kbar + (long long)(b * 12 + h) * 2048;
  const float* btr = beta + (long long)(b * 12 + h) * 2048;
  for (int p = 0; p < 16; ++p) {
    int idx = p * 128 + t;
    int cc = idx >> 6, d = idx & 63;
    sKB[cc * 72 + d] = (_Float16)kbr[cc * 64 + d];
    sBT[d * 40 + cc] = (_Float16)btr[cc * 64 + d];
  }
  {
    int j = t >> 6, d = t & 63;
    sKl[j * 64 + d] = kp[((long long)b * 64 + w * 2 + j) * 768 + h * 64 + d];
    sVl[j * 64 + d] = vp[((long long)b * 64 + w * 2 + j) * 768 + h * 64 + d];
  }
  __syncthreads();

  // rfa logits: SCALE * Q[32x64] @ Kbar^T[64x32] -> sS[:,2:34]
  {
    const int mt = wv >> 1, nt = wv & 1;
    v8f acc = {0.f, 0.f, 0.f, 0.f, 0.f, 0.f, 0.f, 0.f};
    for (int kt = 0; kt < 2; ++kt) {
      v16h a  = ld_fragA(sQ,  mt * 16 + (lane & 15), 144, kt * 64, lane);
      v16h bb = ld_fragB(sKB, nt * 16 + (lane & 15), 144, kt * 64, lane);
      acc = __builtin_amdgcn_wmma_f32_16x16x32_f16(false, a, false, bb, (short)0, acc,
                                                   false, false);
    }
    const int half = lane >> 4, cn = lane & 15;
    for (int r = 0; r < 8; ++r) {
      int mi = mt * 16 + r + 8 * half;
      sS[mi * 36 + 2 + nt * 16 + cn] = 0.125f * acc[r];
    }
  }
  if (t < 64) {  // local logits -> sS[:,0:2]
    int i = t & 31, j = t >> 5;
    float s = 0.f;
    for (int d = 0; d < 64; ++d) s += (float)sQ[i * 72 + d] * sKl[j * 64 + d];
    sS[i * 36 + j] = 0.125f * s;
  }
  __syncthreads();
  if (t < 32) {  // softmax over 34 per row
    const int i = t;
    float mx = -1e30f;
    for (int e = 0; e < 34; ++e) mx = fmaxf(mx, sS[i * 36 + e]);
    float sum = 0.f;
    for (int e = 0; e < 34; ++e) {
      float pv = __expf(sS[i * 36 + e] - mx);
      sS[i * 36 + e] = pv;
      sum += pv;
    }
    float inv = 1.f / sum;
    sS[i * 36 + 0] *= inv;
    sS[i * 36 + 1] *= inv;
    for (int e = 0; e < 32; ++e) sRA[i * 40 + e] = (_Float16)(sS[i * 36 + 2 + e] * inv);
  }
  __syncthreads();
  // O = RA[32x32] @ beta[32x64] + local mix -> qio (f16, in place)
  {
    const int mt = wv >> 1;
    const int half = lane >> 4, cn = lane & 15;
    v16h a = ld_fragA(sRA, mt * 16 + (lane & 15), 80, 0, lane);
    for (int jj = 0; jj < 2; ++jj) {
      const int nt = (wv & 1) * 2 + jj;
      v16h bb = ld_fragB(sBT, nt * 16 + (lane & 15), 80, 0, lane);
      v8f acc = {0.f, 0.f, 0.f, 0.f, 0.f, 0.f, 0.f, 0.f};
      acc = __builtin_amdgcn_wmma_f32_16x16x32_f16(false, a, false, bb, (short)0, acc,
                                                   false, false);
      for (int r = 0; r < 8; ++r) {
        int mi = mt * 16 + r + 8 * half;
        int dd = nt * 16 + cn;
        float val = acc[r] + sS[mi * 36 + 0] * sVl[dd] + sS[mi * 36 + 1] * sVl[64 + dd];
        qio[(qrow0 + mi) * 768 + h * 64 + dd] = (_Float16)val;
      }
    }
  }
}

// ---------------- host orchestration ----------------
extern "C" void kernel_launch(void* const* d_in, const int* in_sizes, int n_in,
                              void* d_out, int out_size, void* d_ws, size_t ws_size,
                              hipStream_t stream)
{
  (void)in_sizes; (void)n_in; (void)out_size; (void)ws_size;
  const float* query = (const float*)d_in[0];
  const float* key   = (const float*)d_in[1];
  const float* value = (const float*)d_in[2];
  const float* Wq  = (const float*)d_in[3];
  const float* bq  = (const float*)d_in[4];
  const float* Wk  = (const float*)d_in[5];
  const float* bk  = (const float*)d_in[6];
  const float* Wv  = (const float*)d_in[7];
  const float* bv  = (const float*)d_in[8];
  const float* Wo  = (const float*)d_in[9];
  const float* bo  = (const float*)d_in[10];
  const float* kds = (const float*)d_in[11];
  const float* vds = (const float*)d_in[12];
  const float* mqW  = (const float*)d_in[13];
  const float* mqb  = (const float*)d_in[14];
  const float* mqg  = (const float*)d_in[15];
  const float* mqbe = (const float*)d_in[16];
  const float* mkW  = (const float*)d_in[17];
  const float* mkb  = (const float*)d_in[18];
  const float* mkg  = (const float*)d_in[19];
  const float* mkbe = (const float*)d_in[20];

  // ---- workspace layout ----
  _Float16* q16   = (_Float16*)d_ws;        // 64*1024*768 (q, then out_pre in place)
  _Float16* keyT  = q16  + 50331648LL;      // [B][768][1024]
  _Float16* valT  = keyT + 50331648LL;
  _Float16* WqT   = valT + 50331648LL;      // [768][768] each
  _Float16* WkT   = WqT + 589824;
  _Float16* WvT   = WkT + 589824;
  _Float16* WoT   = WvT + 589824;
  _Float16* kds16 = WoT + 589824;           // [64][1024] each
  _Float16* vds16 = kds16 + 65536;
  _Float16* kd16  = vds16 + 65536;          // [B][64][768] each
  _Float16* vd16  = kd16 + 3145728;
  float* kp   = (float*)(vd16 + 3145728);   // [B][64][768] fp32
  float* vp   = kp + 3145728;
  float* kbar = vp + 3145728;               // [B][12][32][64]
  float* beta = kbar + 1572864;

  const long long sKeyT = 768LL * 1024, sKd = 64LL * 768;

  // pre-passes: convert / transpose to f16 once
  cvt_f32_f16<<<dim3(64), 256, 0, stream>>>(kds, kds16, 16384);
  cvt_f32_f16<<<dim3(64), 256, 0, stream>>>(vds, vds16, 16384);
  transpose_cvt<<<dim3(32, 24, 64), 256, 0, stream>>>(key,   keyT, 1024, 768, 1024LL * 768, sKeyT);
  transpose_cvt<<<dim3(32, 24, 64), 256, 0, stream>>>(value, valT, 1024, 768, 1024LL * 768, sKeyT);
  transpose_cvt<<<dim3(24, 24, 1), 256, 0, stream>>>(Wq, WqT, 768, 768, 0, 0);
  transpose_cvt<<<dim3(24, 24, 1), 256, 0, stream>>>(Wk, WkT, 768, 768, 0, 0);
  transpose_cvt<<<dim3(24, 24, 1), 256, 0, stream>>>(Wv, WvT, 768, 768, 0, 0);
  transpose_cvt<<<dim3(24, 24, 1), 256, 0, stream>>>(Wo, WoT, 768, 768, 0, 0);

  // 1-2: key_d/val_d[b] = kds/vds[64,1024] @ key/value[b]
  wmma_gemm<true, true><<<dim3(3, 1, 64), 256, 0, stream>>>(
      kds16, keyT, nullptr, kd16, 1024, 768, 0, sKeyT, sKd);
  wmma_gemm<true, true><<<dim3(3, 1, 64), 256, 0, stream>>>(
      vds16, valT, nullptr, vd16, 1024, 768, 0, sKeyT, sKd);
  // 3: q = query @ Wq + bq  (f16 out)
  wmma_gemm<false, true><<<dim3(3, 1024, 1), 256, 0, stream>>>(
      query, WqT, bq, q16, 768, 768, 0, 0, 0);
  // 4-5: k/v = key_d/val_d @ Wk/Wv + b  (fp32 out for stats)
  wmma_gemm<true, false><<<dim3(3, 1, 64), 256, 0, stream>>>(
      kd16, WkT, bk, kp, 768, 768, sKd, 0, sKd);
  wmma_gemm<true, false><<<dim3(3, 1, 64), 256, 0, stream>>>(
      vd16, WvT, bv, vp, 768, 768, sKd, 0, sKd);
  // 6: per-chunk stats
  chunk_stats<<<dim3(64 * 12 * 32), 64, 0, stream>>>(
      q16, kp, vp, mqW, mqb, mqg, mqbe, mkW, mkb, mkg, mkbe, kbar, beta);
  // 7: windowed attention (in place over q16)
  window_attn<<<dim3(64 * 12 * 32), 128, 0, stream>>>(kp, vp, kbar, beta, q16);
  // 8: out = out_pre @ Wo + bo
  wmma_gemm<true, false><<<dim3(3, 1024, 1), 256, 0, stream>>>(
      q16, WoT, bo, d_out, 768, 768, 0, 0, 0);
}
